// DeepClusteringNet_71700184039604
// MI455X (gfx1250) — matrix operand order repaired and checked by reading, not compiled, not run on hardware
//
#include <hip/hip_runtime.h>

typedef __attribute__((ext_vector_type(4)))  __bf16 v4bf;
typedef __attribute__((ext_vector_type(8)))  __bf16 v8bf;
typedef __attribute__((ext_vector_type(16))) __bf16 v16bf;
typedef __attribute__((ext_vector_type(4)))  float  v4f;
typedef __attribute__((ext_vector_type(8)))  float  v8f;
typedef __attribute__((ext_vector_type(4)))  int    v4i;

// Problem dims (fixed by the reference)
#define BB 8192
#define DD 2048
#define H1 1024
#define H2 512
#define LL 256

#define BM 128
#define BN 256
#define BK 32
#define LDS_STRIDE (BK + 8)   // 40 bf16 = 80 B rows: keeps 16B chunks aligned

typedef __attribute__((address_space(1))) v4i as1_v4i;
typedef __attribute__((address_space(3))) v4i as3_v4i;

// ---- async global->LDS 16B copy (CDNA5: tracked on ASYNCcnt) ----
__device__ __forceinline__ void async_copy_b128(void* lptr, const void* gptr) {
#if __has_builtin(__builtin_amdgcn_global_load_async_to_lds_b128)
    __builtin_amdgcn_global_load_async_to_lds_b128(
        (as1_v4i*)gptr, (as3_v4i*)lptr, 0, 0);
#else
    unsigned lds_off = (unsigned)(unsigned long long)(as3_v4i*)lptr;
    asm volatile("global_load_async_to_lds_b128 %0, %1, off"
                 :: "v"(lds_off), "v"((unsigned long long)gptr)
                 : "memory");
#endif
}
__device__ __forceinline__ void wait_async0() {
#if __has_builtin(__builtin_amdgcn_s_wait_asynccnt)
    __builtin_amdgcn_s_wait_asynccnt(0);
#else
    asm volatile("s_wait_asynccnt 0x0" ::: "memory");
#endif
}

// ---------------------------------------------------------------------------
// One-time weight prep: W[K][N] f32  ->  Wt[N][K] bf16 (tiled LDS transpose)
// ---------------------------------------------------------------------------
__global__ __launch_bounds__(256)
void transpose_w_kernel(const float* __restrict__ W, __bf16* __restrict__ Wt,
                        int K, int N)
{
    __shared__ float tile[32][33];
    int bn = blockIdx.x * 32;            // n base
    int bk = blockIdx.y * 32;            // k base
    int tx = threadIdx.x & 31;
    int ty = threadIdx.x >> 5;           // 0..7
    #pragma unroll
    for (int q = 0; q < 4; ++q)
        tile[ty + 8 * q][tx] = W[(size_t)(bk + ty + 8 * q) * (size_t)N + (size_t)(bn + tx)];
    __syncthreads();
    #pragma unroll
    for (int q = 0; q < 4; ++q)
        Wt[(size_t)(bn + ty + 8 * q) * (size_t)K + (size_t)(bk + tx)] =
            (__bf16)tile[tx][ty + 8 * q];
}

// ---------------------------------------------------------------------------
// Tiled bf16-WMMA GEMM:  C[M,N] = act(A'[M,K] @ Wt^T + bias),  Wt bf16 [N][K]
// MODE: 0 = ReLU + store act ; 1 = ReLU + store act + BN stats ; 2 = loss vs X
// K, N compile-time so address math folds to immediate offsets.
// ---------------------------------------------------------------------------
template <int A_BF16, int HAS_AFFINE, int MODE, int K, int N>
__global__ __launch_bounds__(256)
void gemm_wmma(const void* __restrict__ Ap,
               const float* __restrict__ scaleA, const float* __restrict__ shiftA,
               const __bf16* __restrict__ Wt, const float* __restrict__ bias,
               __bf16* __restrict__ outAct,
               float* __restrict__ colSum, float* __restrict__ colSumSq,
               const float* __restrict__ Xref, float* __restrict__ lossAcc,
               float lossCoef)
{
    __shared__ __bf16 As [BM][LDS_STRIDE];   // [row][k]
    __shared__ __bf16 Bst[BN][LDS_STRIDE];   // [col][k]

    const int tid   = threadIdx.x;
    const int lane  = tid & 31;
    const int wave  = tid >> 5;
    const int wm0   = (wave >> 2) * 64;   // 2 waves down   -> 128 rows
    const int wn0   = (wave &  3) * 64;   // 4 waves across -> 256 cols
    const int lrow  = lane & 15;
    const int khalf = lane >> 4;

    const int blockM0 = blockIdx.y * BM;
    const int blockN0 = blockIdx.x * BN;

    v8f acc[4][4] = {};

    for (int k0 = 0; k0 < K; k0 += BK) {
        // ---------------- stage A tile ----------------
        if (A_BF16) {
            const __bf16* A = (const __bf16*)Ap;
            #pragma unroll
            for (int i = 0; i < 2; ++i) {
                int flat = tid + 256 * i;          // 512 slots of 8 bf16
                int r  = flat >> 2;
                int ko = (flat & 3) * 8;
                const __bf16* src = A + (size_t)(blockM0 + r) * K + (size_t)(k0 + ko);
                if (HAS_AFFINE) {
                    v8bf v = *(const v8bf*)src;
                    v4f s0 = *(const v4f*)&scaleA[k0 + ko];
                    v4f s1 = *(const v4f*)&scaleA[k0 + ko + 4];
                    v4f t0 = *(const v4f*)&shiftA[k0 + ko];
                    v4f t1 = *(const v4f*)&shiftA[k0 + ko + 4];
                    #pragma unroll
                    for (int q = 0; q < 4; ++q) {
                        v[q]     = (__bf16)((float)v[q]     * s0[q] + t0[q]);
                        v[q + 4] = (__bf16)((float)v[q + 4] * s1[q] + t1[q]);
                    }
                    *(v8bf*)&As[r][ko] = v;
                } else {
                    async_copy_b128(&As[r][ko], src);   // pure copy: DMA past VGPRs
                }
            }
        } else {
            const float* A = (const float*)Ap;
            #pragma unroll
            for (int i = 0; i < 4; ++i) {
                int flat = tid + 256 * i;          // 1024 float4 slots
                int r  = flat >> 3;
                int kq = (flat & 7) * 4;
                v4f v = *(const v4f*)(A + (size_t)(blockM0 + r) * K + (size_t)(k0 + kq));
                v4bf c;
                #pragma unroll
                for (int q = 0; q < 4; ++q) c[q] = (__bf16)v[q];
                *(v4bf*)&As[r][kq] = c;
            }
        }
        // ---------------- stage B tile: async DMA of pre-transposed bf16 weights ----
        #pragma unroll
        for (int i = 0; i < 4; ++i) {
            int flat = tid + 256 * i;              // 1024 slots of 8 bf16 (256 cols)
            int r  = flat >> 2;
            int ko = (flat & 3) * 8;
            async_copy_b128(&Bst[r][ko],
                            Wt + (size_t)(blockN0 + r) * K + (size_t)(k0 + ko));
        }
        wait_async0();
        __syncthreads();

        // ---------------- fragments (contiguous 16B LDS chunks) ----------------
        v16bf aF[4], bF[4];
        #pragma unroll
        for (int tm = 0; tm < 4; ++tm) {
            const __bf16* rp = &As[wm0 + tm * 16 + lrow][0];
            v8bf lo = *(const v8bf*)(rp + 8 * khalf);        // K = 8*khalf .. +7
            v8bf hi = *(const v8bf*)(rp + 16 + 8 * khalf);   // K = 16+8*khalf .. +7
            aF[tm] = __builtin_shufflevector(lo, hi, 0,1,2,3,4,5,6,7,8,9,10,11,12,13,14,15);
        }
        #pragma unroll
        for (int tn = 0; tn < 4; ++tn) {
            const __bf16* cp = &Bst[wn0 + tn * 16 + lrow][8 * khalf];  // K = 8*khalf .. +7, 16+8*khalf .. +7
            v8bf lo = *(const v8bf*)cp;
            v8bf hi = *(const v8bf*)(cp + 16);
            bF[tn] = __builtin_shufflevector(lo, hi, 0,1,2,3,4,5,6,7,8,9,10,11,12,13,14,15);
        }

        // ---------------- 16 WMMAs / wave / K-step ----------------
        #pragma unroll
        for (int tm = 0; tm < 4; ++tm)
            #pragma unroll
            for (int tn = 0; tn < 4; ++tn)
                acc[tm][tn] = __builtin_amdgcn_wmma_f32_16x16x32_bf16(
                    false, aF[tm], false, bF[tn],
                    (short)0, acc[tm][tn], false, false);

        __syncthreads();
    }

    // ---------------- epilogue ----------------
    float lloss = 0.f;
    #pragma unroll
    for (int tm = 0; tm < 4; ++tm) {
        #pragma unroll
        for (int tn = 0; tn < 4; ++tn) {
            int gcol = blockN0 + wn0 + tn * 16 + lrow;
            int row0 = blockM0 + wm0 + tm * 16 + 8 * khalf;
            float bv = bias[gcol];
            float csum = 0.f, csumsq = 0.f;
            if (MODE == 2) {
                const float* xp = Xref + (size_t)row0 * N + gcol;
                #pragma unroll
                for (int r = 0; r < 8; ++r) {
                    float v = acc[tm][tn][r] + bv;
                    float d = xp[(size_t)r * N] - v;
                    lloss += d * d;
                }
            } else {
                __bf16* op = outAct + (size_t)row0 * N + gcol;
                #pragma unroll
                for (int r = 0; r < 8; ++r) {
                    float v = acc[tm][tn][r] + bv;
                    v = v > 0.f ? v : 0.f;  // ReLU
                    op[(size_t)r * N] = (__bf16)v;
                    if (MODE == 1) { csum += v; csumsq += v * v; }
                }
            }
            if (MODE == 1) {
                atomicAdd(&colSum[gcol], csum);
                atomicAdd(&colSumSq[gcol], csumsq);
            }
        }
    }
    if (MODE == 2) atomicAdd(lossAcc, lloss * lossCoef);
}

// scale[j] = gamma[j]/sqrt(var+eps); shift[j] = beta[j] - mean[j]*scale[j]
__global__ void bn_scale_kernel(const float* __restrict__ sum,
                                const float* __restrict__ sumsq,
                                const float* __restrict__ g,
                                const float* __restrict__ be,
                                float* __restrict__ scale,
                                float* __restrict__ shift,
                                int C, float invB)
{
    int j = blockIdx.x * blockDim.x + threadIdx.x;
    if (j < C) {
        float m   = sum[j] * invB;
        float var = sumsq[j] * invB - m * m;
        float s   = g[j] * rsqrtf(var + 1e-5f);
        scale[j] = s;
        shift[j] = be[j] - m * s;
    }
}

// dist_loss = 0.5*beta * sum_i || BN(relu_latent_i) - clusters[cid_i] ||^2
__global__ __launch_bounds__(256)
void dist_loss_kernel(const __bf16* __restrict__ act3,
                      const float* __restrict__ scale,
                      const float* __restrict__ shift,
                      const float* __restrict__ clusters,
                      const int* __restrict__ cid,
                      float* __restrict__ lossAcc)
{
    __shared__ float red[256];
    const size_t nOct = (size_t)BB * (size_t)LL / 8;
    float local = 0.f;
    for (size_t p = (size_t)blockIdx.x * blockDim.x + threadIdx.x;
         p < nOct; p += (size_t)gridDim.x * blockDim.x) {
        size_t e = p * 8;
        int i = (int)(e >> 8);         // L = 256
        int j = (int)(e & 255);
        v8bf a = *(const v8bf*)(act3 + e);
        const float* cl = &clusters[(size_t)cid[i] * LL + j];
        #pragma unroll
        for (int q = 0; q < 8; ++q) {
            float lat = (float)a[q] * scale[j + q] + shift[j + q];
            float d = lat - cl[q];
            local += d * d;
        }
    }
    red[threadIdx.x] = local;
    __syncthreads();
    for (int s = 128; s > 0; s >>= 1) {
        if (threadIdx.x < s) red[threadIdx.x] += red[threadIdx.x + s];
        __syncthreads();
    }
    if (threadIdx.x == 0) atomicAdd(lossAcc, 0.5f * 1.0f * red[0]); // BETA=1
}

// ---------------------------------------------------------------------------
// Workspace layout (bytes)
// ---------------------------------------------------------------------------
#define ACT1_OFF 0ull
#define ACT1_BYTES ((size_t)BB * H1 * 2)
#define ACT2_OFF (ACT1_OFF + ACT1_BYTES)
#define ACT2_BYTES ((size_t)BB * H2 * 2)
#define ACT3_OFF (ACT2_OFF + ACT2_BYTES)
#define ACT3_BYTES ((size_t)BB * LL * 2)
#define ACT4_OFF (ACT3_OFF + ACT3_BYTES)
#define ACT4_BYTES ((size_t)BB * H2 * 2)
#define ACT5_OFF (ACT4_OFF + ACT4_BYTES)
#define ACT5_BYTES ((size_t)BB * H1 * 2)
#define STATS_OFF (ACT5_OFF + ACT5_BYTES)
#define STATS_FLOATS (4 * (H2 + LL + H2 + H1))
#define WT_OFF (STATS_OFF + ((STATS_FLOATS * 4 + 255) & ~255))
// transposed bf16 weights, in chain order
#define WT0_ELEMS ((size_t)DD * H1)
#define WT1_ELEMS ((size_t)H1 * H2)
#define WT2_ELEMS ((size_t)H2 * LL)
#define WT3_ELEMS ((size_t)LL * H2)
#define WT4_ELEMS ((size_t)H2 * H1)
#define WT5_ELEMS ((size_t)H1 * DD)

extern "C" void kernel_launch(void* const* d_in, const int* in_sizes, int n_in,
                              void* d_out, int out_size, void* d_ws, size_t ws_size,
                              hipStream_t stream) {
    const float* X        = (const float*)d_in[0];
    const int*   cid      = (const int*)  d_in[1];
    const float* clusters = (const float*)d_in[2];
    const float* enc_w0 = (const float*)d_in[3];  const float* enc_b0 = (const float*)d_in[4];
    const float* enc_w1 = (const float*)d_in[5];  const float* enc_b1 = (const float*)d_in[6];
    const float* enc_g1 = (const float*)d_in[7];  const float* enc_be1= (const float*)d_in[8];
    const float* enc_w2 = (const float*)d_in[9];  const float* enc_b2 = (const float*)d_in[10];
    const float* enc_g2 = (const float*)d_in[11]; const float* enc_be2= (const float*)d_in[12];
    const float* dec_w0 = (const float*)d_in[13]; const float* dec_b0 = (const float*)d_in[14];
    const float* dec_g0 = (const float*)d_in[15]; const float* dec_be0= (const float*)d_in[16];
    const float* dec_w1 = (const float*)d_in[17]; const float* dec_b1 = (const float*)d_in[18];
    const float* dec_g1 = (const float*)d_in[19]; const float* dec_be1= (const float*)d_in[20];
    const float* dec_w2 = (const float*)d_in[21]; const float* dec_b2 = (const float*)d_in[22];

    float* out = (float*)d_out;
    char*  ws  = (char*)d_ws;

    __bf16* act1 = (__bf16*)(ws + ACT1_OFF);
    __bf16* act2 = (__bf16*)(ws + ACT2_OFF);
    __bf16* act3 = (__bf16*)(ws + ACT3_OFF);
    __bf16* act4 = (__bf16*)(ws + ACT4_OFF);
    __bf16* act5 = (__bf16*)(ws + ACT5_OFF);
    float*  st   = (float*)(ws + STATS_OFF);

    float* sum2   = st;              float* sumsq2 = sum2 + H2;
    float* scale2 = sumsq2 + H2;     float* shift2 = scale2 + H2;
    float* sum3   = shift2 + H2;     float* sumsq3 = sum3 + LL;
    float* scale3 = sumsq3 + LL;     float* shift3 = scale3 + LL;
    float* sum4   = shift3 + LL;     float* sumsq4 = sum4 + H2;
    float* scale4 = sumsq4 + H2;     float* shift4 = scale4 + H2;
    float* sum5   = shift4 + H2;     float* sumsq5 = sum5 + H1;
    float* scale5 = sumsq5 + H1;     float* shift5 = scale5 + H1;

    __bf16* wt0 = (__bf16*)(ws + WT_OFF);
    __bf16* wt1 = wt0 + WT0_ELEMS;
    __bf16* wt2 = wt1 + WT1_ELEMS;
    __bf16* wt3 = wt2 + WT2_ELEMS;
    __bf16* wt4 = wt3 + WT3_ELEMS;
    __bf16* wt5 = wt4 + WT4_ELEMS;

    (void)hipMemsetAsync(out, 0, sizeof(float), stream);
    (void)hipMemsetAsync(st, 0, STATS_FLOATS * sizeof(float), stream);

    dim3 blk(256);

    // one-time (per launch) weight transpose + bf16 convert
    transpose_w_kernel<<<dim3(H1 / 32, DD / 32), blk, 0, stream>>>(enc_w0, wt0, DD, H1);
    transpose_w_kernel<<<dim3(H2 / 32, H1 / 32), blk, 0, stream>>>(enc_w1, wt1, H1, H2);
    transpose_w_kernel<<<dim3(LL / 32, H2 / 32), blk, 0, stream>>>(enc_w2, wt2, H2, LL);
    transpose_w_kernel<<<dim3(H2 / 32, LL / 32), blk, 0, stream>>>(dec_w0, wt3, LL, H2);
    transpose_w_kernel<<<dim3(H1 / 32, H2 / 32), blk, 0, stream>>>(dec_w1, wt4, H2, H1);
    transpose_w_kernel<<<dim3(DD / 32, H1 / 32), blk, 0, stream>>>(dec_w2, wt5, H1, DD);

    const float invB = 1.0f / (float)BB;
    const float recCoef = 0.1f / ((float)BB * (float)DD);   // LAMDA * mean

    // enc layer 0: f32 A, ReLU only
    gemm_wmma<0,0,0,DD,H1><<<dim3(H1 / BN, BB / BM), blk, 0, stream>>>(
        X, nullptr, nullptr, wt0, enc_b0,
        act1, nullptr, nullptr, nullptr, nullptr, 0.f);

    // enc layer 1: bf16 A (async-staged), ReLU + BN stats
    gemm_wmma<1,0,1,H1,H2><<<dim3(H2 / BN, BB / BM), blk, 0, stream>>>(
        act1, nullptr, nullptr, wt1, enc_b1,
        act2, sum2, sumsq2, nullptr, nullptr, 0.f);
    bn_scale_kernel<<<(H2 + 255) / 256, 256, 0, stream>>>(
        sum2, sumsq2, enc_g1, enc_be1, scale2, shift2, H2, invB);

    // enc layer 2 (latent): BN fold on A, ReLU + BN stats
    gemm_wmma<1,1,1,H2,LL><<<dim3(LL / BN, BB / BM), blk, 0, stream>>>(
        act2, scale2, shift2, wt2, enc_b2,
        act3, sum3, sumsq3, nullptr, nullptr, 0.f);
    bn_scale_kernel<<<(LL + 255) / 256, 256, 0, stream>>>(
        sum3, sumsq3, enc_g2, enc_be2, scale3, shift3, LL, invB);

    // distance loss on normalized latent
    dist_loss_kernel<<<2048, 256, 0, stream>>>(act3, scale3, shift3, clusters, cid, out);

    // dec layer 0
    gemm_wmma<1,1,1,LL,H2><<<dim3(H2 / BN, BB / BM), blk, 0, stream>>>(
        act3, scale3, shift3, wt3, dec_b0,
        act4, sum4, sumsq4, nullptr, nullptr, 0.f);
    bn_scale_kernel<<<(H2 + 255) / 256, 256, 0, stream>>>(
        sum4, sumsq4, dec_g0, dec_be0, scale4, shift4, H2, invB);

    // dec layer 1
    gemm_wmma<1,1,1,H2,H1><<<dim3(H1 / BN, BB / BM), blk, 0, stream>>>(
        act4, scale4, shift4, wt4, dec_b1,
        act5, sum5, sumsq5, nullptr, nullptr, 0.f);
    bn_scale_kernel<<<(H1 + 255) / 256, 256, 0, stream>>>(
        sum5, sumsq5, dec_g1, dec_be1, scale5, shift5, H1, invB);

    // dec layer 2: fused reconstruction loss vs X (no rec_X materialization)
    gemm_wmma<1,1,2,H1,DD><<<dim3(DD / BN, BB / BM), blk, 0, stream>>>(
        act5, scale5, shift5, wt5, dec_b2,
        nullptr, nullptr, nullptr, X, out, recCoef);
}
